// SparseMoE_6536940224818
// MI455X (gfx1250) — matrix operand order repaired and checked by reading, not compiled, not run on hardware
//
#include <hip/hip_runtime.h>
#include <hip/hip_bf16.h>
#include <math.h>

// Problem constants (match reference)
#define E_    8
#define T_    8192          // B*S
#define H_    1024
#define I_    4096
#define TILE_M 32           // tokens per block (2 WMMA m-tiles: B reuse x2)
#define MT    2             // m-tiles per block
#define ICH    512          // I-chunk held in LDS per pass
#define H_PAD  1032         // 1024 + 8  (stride = 516 dwords = 4 mod 64 banks)
#define ICH_PAD 520         // 512 + 8   (stride = 260 dwords = 4 mod 64 banks)

// dynamic LDS layout (bytes)
#define XBF_OFS  0
#define XBF_SZ   (TILE_M * H_PAD * 2)          // 66048
#define HBF_OFS  (XBF_OFS + XBF_SZ)
#define HBF_SZ   (TILE_M * ICH_PAD * 2)        // 33280
#define TOK_OFS  (HBF_OFS + HBF_SZ)
#define WT_OFS   (TOK_OFS + TILE_M * 4)
#define SMEM_BYTES (WT_OFS + TILE_M * 4)       // 99584

typedef __attribute__((ext_vector_type(16))) __bf16         v16bf;
typedef __attribute__((ext_vector_type(4)))  __bf16         bf4;
typedef __attribute__((ext_vector_type(8)))  float          v8f;
typedef __attribute__((ext_vector_type(16))) float          vf16;
typedef __attribute__((ext_vector_type(8)))  unsigned short us8;
typedef __attribute__((ext_vector_type(4)))  float          f4;

// A fragment (A 16x32; contiguous-K order used here, not the ISA table order): lane holds row m=lane&15,
// K = 16*(lane>>4) + j  -> 16 contiguous K values per lane (two 16B loads).
__device__ __forceinline__ v16bf load_a_frag_lds(const __bf16* base) {
    union { us8 h[2]; v16bf v; } fr;
    fr.h[0] = *(const us8*)(base);
    fr.h[1] = *(const us8*)(base + 8);
    return fr.v;
}

// B fragment (B 32x16): lane holds column n=lane&15, K = 16*(lane>>4) + j
// -> 16 contiguous K values per lane.
__device__ __forceinline__ v16bf load_b_frag_bf16(const __bf16* __restrict__ p) {
    union { us8 h[2]; v16bf v; } fr;
    fr.h[0] = *(const us8*)(p);
    fr.h[1] = *(const us8*)(p + 8);
    return fr.v;
}

// Fallback: fp32 source, packed hw convert (v_cvt_pk_bf16_f32).
__device__ __forceinline__ v16bf load_b_frag_f32(const float* __restrict__ p) {
    union { f4 q[4]; vf16 f; } t;
    t.q[0] = *(const f4*)(p);
    t.q[1] = *(const f4*)(p + 4);
    t.q[2] = *(const f4*)(p + 8);
    t.q[3] = *(const f4*)(p + 12);
    return __builtin_convertvector(t.f, v16bf);
}

__device__ __forceinline__ float gelu_exact(float v) {
    return 0.5f * v * (1.0f + erff(v * 0.70710678118654752f));
}

// ---------------------------------------------------------------- kernel 0
__global__ __launch_bounds__(256) void moe_zero_kernel(float* __restrict__ out,
                                                       int* __restrict__ cnt) {
    size_t i = (size_t)blockIdx.x * blockDim.x + threadIdx.x;
    const size_t n = (size_t)T_ * H_;
    const size_t stride = (size_t)gridDim.x * blockDim.x;
    for (size_t k = i; k < n; k += stride) out[k] = 0.0f;
    if (i < E_) cnt[i] = 0;
}

// ---------------------------------------------------------------- kernel 0b
// One-time fp32 -> bf16 weight conversion (memory bound, ~400 MB ≈ 17 us);
// removes converts from the WMMA loops and halves weight-stream bytes.
__global__ __launch_bounds__(256) void moe_convert_kernel(
    const float* __restrict__ wfc, const float* __restrict__ wpj,
    __bf16* __restrict__ ofc, __bf16* __restrict__ opj) {
    const size_t n4 = (size_t)E_ * I_ * H_ / 4;
    const size_t stride = (size_t)gridDim.x * blockDim.x;
    for (size_t k = (size_t)blockIdx.x * blockDim.x + threadIdx.x; k < n4; k += stride) {
        f4 a = ((const f4*)wfc)[k];
        f4 b = ((const f4*)wpj)[k];
        ((bf4*)ofc)[k] = __builtin_convertvector(a, bf4);
        ((bf4*)opj)[k] = __builtin_convertvector(b, bf4);
    }
}

// ---------------------------------------------------------------- kernel 1
// One wave32 per token: 8 routing dots, softmax, top-2, renorm; build lists.
__global__ __launch_bounds__(256) void moe_router_kernel(
    const float* __restrict__ x, const float* __restrict__ w_gate,
    float* __restrict__ logits, int* __restrict__ cnt,
    int* __restrict__ tlist, float* __restrict__ wlist) {
    const int wave = threadIdx.x >> 5;
    const int lane = threadIdx.x & 31;
    const int t = blockIdx.x * 8 + wave;

    float acc[E_];
#pragma unroll
    for (int e = 0; e < E_; ++e) acc[e] = 0.0f;

    const float* xr = x + (size_t)t * H_;
    for (int h = lane; h < H_; h += 32) {
        float xv = xr[h];
#pragma unroll
        for (int e = 0; e < E_; ++e) acc[e] += xv * w_gate[e * H_ + h];
    }
#pragma unroll
    for (int e = 0; e < E_; ++e)
        for (int off = 16; off > 0; off >>= 1) acc[e] += __shfl_down(acc[e], off);

    if (lane == 0) {
        float mx = acc[0];
#pragma unroll
        for (int e = 1; e < E_; ++e) mx = fmaxf(mx, acc[e]);
        float p[E_];
#pragma unroll
        for (int e = 0; e < E_; ++e) p[e] = expf(acc[e] - mx);
#pragma unroll
        for (int e = 0; e < E_; ++e) logits[(size_t)t * E_ + e] = acc[e];
        // top-2 (ties -> lowest index, matching lax.top_k)
        int e0 = 0;
#pragma unroll
        for (int e = 1; e < E_; ++e) if (p[e] > p[e0]) e0 = e;
        int e1 = (e0 == 0) ? 1 : 0;
#pragma unroll
        for (int e = 0; e < E_; ++e) if (e != e0 && p[e] > p[e1]) e1 = e;
        float w0 = p[e0], w1 = p[e1];
        float inv = 1.0f / (w0 + w1);       // softmax denom cancels in renorm
        w0 *= inv; w1 *= inv;
        int s0 = atomicAdd(&cnt[e0], 1);
        tlist[e0 * T_ + s0] = t; wlist[e0 * T_ + s0] = w0;
        int s1 = atomicAdd(&cnt[e1], 1);
        tlist[e1 * T_ + s1] = t; wlist[e1 * T_ + s1] = w1;
    }
}

// ---------------------------------------------------------------- kernel 2
// One block = (expert, tile of 32 gathered tokens). Fused fc+GELU+proj.
// Each B fragment feeds MT=2 WMMAs -> halves weight traffic per FLOP.
template <bool PRE>
__global__ __launch_bounds__(256) void moe_expert_kernel(
    const float* __restrict__ x,
    const float* __restrict__ w_fc_f32,  const __bf16* __restrict__ w_fc_bf,
    const float* __restrict__ b_fc,
    const float* __restrict__ w_proj_f32, const __bf16* __restrict__ w_proj_bf,
    const float* __restrict__ b_proj, const int* __restrict__ cnt,
    const int* __restrict__ tlist, const float* __restrict__ wlist,
    float* __restrict__ out) {
    const int e = blockIdx.y;
    const int n_tok = cnt[e];
    const int tile0 = (int)blockIdx.x * TILE_M;
    if (tile0 >= n_tok) return;                 // uniform early exit

    extern __shared__ __align__(16) char smem[];
    __bf16* Xbf  = (__bf16*)(smem + XBF_OFS);   // [TILE_M][H_PAD]
    __bf16* Hbf  = (__bf16*)(smem + HBF_OFS);   // [TILE_M][ICH_PAD]
    int*    s_tok = (int*)(smem + TOK_OFS);
    float*  s_wt  = (float*)(smem + WT_OFS);

    const int tid  = threadIdx.x;
    const int wave = tid >> 5;
    const int lane = tid & 31;
    const int g    = lane >> 4;     // half-wave select
    const int ln   = lane & 15;

    if (tid < TILE_M) {
        int idx = tile0 + tid;
        if (idx < n_tok) {
            s_tok[tid] = tlist[e * T_ + idx];
            s_wt[tid]  = wlist[e * T_ + idx];
        } else {
            s_tok[tid] = 0;
            s_wt[tid]  = 0.0f;
        }
    }
    __syncthreads();

    // Gather 32 token rows, fp32 -> bf16 into LDS (packed hw converts).
    {
        const int row = tid >> 3;               // 0..31
        const int cb  = (tid & 7) * 128;        // 128 floats per thread
        const float* src = x + (size_t)s_tok[row] * H_ + cb;
#pragma unroll
        for (int k = 0; k < 32; ++k) {
            f4 v = *(const f4*)(src + k * 4);
            *(bf4*)&Xbf[row * H_PAD + cb + k * 4] = __builtin_convertvector(v, bf4);
        }
    }
    __syncthreads();

    const float*  wfc_f = w_fc_f32   + (size_t)e * I_ * H_;
    const float*  wpj_f = w_proj_f32 + (size_t)e * H_ * I_;
    const __bf16* wfc_b = w_fc_bf    + (size_t)e * I_ * H_;
    const __bf16* wpj_b = w_proj_bf  + (size_t)e * H_ * I_;

    v8f cout[MT][8];                            // persistent proj accumulators
#pragma unroll
    for (int mt = 0; mt < MT; ++mt)
#pragma unroll
        for (int i = 0; i < 8; ++i)
#pragma unroll
            for (int j = 0; j < 8; ++j) cout[mt][i][j] = 0.0f;

    for (int io = 0; io < I_ / ICH; ++io) {
        // ---- FC: this wave computes ht[0:32, io*ICH + wave*64 .. +64)
        v8f cfc[MT][4];
#pragma unroll
        for (int mt = 0; mt < MT; ++mt)
#pragma unroll
            for (int i = 0; i < 4; ++i)
#pragma unroll
                for (int j = 0; j < 8; ++j) cfc[mt][i][j] = 0.0f;

        const int colbase = io * ICH + wave * 64;
        for (int kc = 0; kc < H_; kc += 32) {
            v16bf a[MT];
#pragma unroll
            for (int mt = 0; mt < MT; ++mt)
                a[mt] = load_a_frag_lds(&Xbf[(mt * 16 + ln) * H_PAD + kc + 16 * g]);
            if (kc + 32 < H_) {
                if (PRE) __builtin_prefetch(wfc_b + (size_t)(colbase + ln) * H_ + kc + 32 + 16 * g, 0, 1);
                else     __builtin_prefetch(wfc_f + (size_t)(colbase + ln) * H_ + kc + 32 + 16 * g, 0, 1);
            }
#pragma unroll
            for (int nt = 0; nt < 4; ++nt) {
                const size_t off = (size_t)(colbase + nt * 16 + ln) * H_ + kc + 16 * g;
                v16bf b = PRE ? load_b_frag_bf16(wfc_b + off)
                              : load_b_frag_f32(wfc_f + off);
#pragma unroll
                for (int mt = 0; mt < MT; ++mt)
                    cfc[mt][nt] = __builtin_amdgcn_wmma_f32_16x16x32_bf16(
                        false, a[mt], false, b, (short)0, cfc[mt][nt], false, false);
            }
        }
        // bias + exact GELU, store bf16 chunk of ht
#pragma unroll
        for (int nt = 0; nt < 4; ++nt) {
            const int col  = colbase + nt * 16 + ln;
            const float bi = b_fc[(size_t)e * I_ + col];
            const int crel = wave * 64 + nt * 16 + ln;
#pragma unroll
            for (int mt = 0; mt < MT; ++mt)
#pragma unroll
                for (int r = 0; r < 8; ++r) {
                    const int m = mt * 16 + r + 8 * g;
                    Hbf[m * ICH_PAD + crel] = (__bf16)gelu_exact(cfc[mt][nt][r] + bi);
                }
        }
        __syncthreads();

        // ---- PROJ partial: wave owns out columns [wave*128, +128)
        for (int kc = 0; kc < ICH; kc += 32) {
            v16bf a[MT];
#pragma unroll
            for (int mt = 0; mt < MT; ++mt)
                a[mt] = load_a_frag_lds(&Hbf[(mt * 16 + ln) * ICH_PAD + kc + 16 * g]);
#pragma unroll
            for (int nt = 0; nt < 8; ++nt) {
                const int hcol = wave * 128 + nt * 16 + ln;
                const size_t off = (size_t)hcol * I_ + io * ICH + kc + 16 * g;
                v16bf b = PRE ? load_b_frag_bf16(wpj_b + off)
                              : load_b_frag_f32(wpj_f + off);
#pragma unroll
                for (int mt = 0; mt < MT; ++mt)
                    cout[mt][nt] = __builtin_amdgcn_wmma_f32_16x16x32_bf16(
                        false, a[mt], false, b, (short)0, cout[mt][nt], false, false);
            }
        }
        __syncthreads();   // before next pass overwrites Hbf
    }

    // ---- scatter: out[token] += weight * (acc + b_proj)   (2 fp32 atomics
    // per element total across experts -> commutative, deterministic)
#pragma unroll
    for (int nt = 0; nt < 8; ++nt) {
        const int hcol = wave * 128 + nt * 16 + ln;
        const float bi = b_proj[(size_t)e * H_ + hcol];
#pragma unroll
        for (int mt = 0; mt < MT; ++mt)
#pragma unroll
            for (int r = 0; r < 8; ++r) {
                const int m = mt * 16 + r + 8 * g;
                if (tile0 + m < n_tok) {
                    atomicAdd(&out[(size_t)s_tok[m] * H_ + hcol],
                              s_wt[m] * (cout[mt][nt][r] + bi));
                }
            }
    }
}

// ---------------------------------------------------------------- launch
extern "C" void kernel_launch(void* const* d_in, const int* in_sizes, int n_in,
                              void* d_out, int out_size, void* d_ws, size_t ws_size,
                              hipStream_t stream) {
    const float* x      = (const float*)d_in[0];
    const float* w_gate = (const float*)d_in[1];
    const float* w_fc   = (const float*)d_in[2];
    const float* b_fc   = (const float*)d_in[3];
    const float* w_proj = (const float*)d_in[4];
    const float* b_proj = (const float*)d_in[5];

    float* out    = (float*)d_out;
    float* logits = out + (size_t)T_ * H_;      // tuple element 2, flat concat

    char*  ws    = (char*)d_ws;
    int*   cnt   = (int*)ws;                                     // E counters
    int*   tlist = (int*)(ws + 256);                             // E*T ints
    float* wlist = (float*)(ws + 256 + sizeof(int) * E_ * T_);   // E*T floats

    const size_t list_end = 256 + (size_t)8 * E_ * T_;
    const size_t wofs     = (list_end + 255) & ~(size_t)255;
    const size_t wbytes   = (size_t)E_ * I_ * H_ * sizeof(unsigned short);
    const bool   pre      = ws_size >= wofs + 2 * wbytes;
    __bf16* wfc_bf = (__bf16*)(ws + wofs);
    __bf16* wpj_bf = (__bf16*)(ws + wofs + wbytes);

    // Allow >64KB dynamic LDS (WGP has 320KB). Host-side, capture-safe.
    hipFuncSetAttribute((const void*)moe_expert_kernel<true>,
                        hipFuncAttributeMaxDynamicSharedMemorySize, SMEM_BYTES);
    hipFuncSetAttribute((const void*)moe_expert_kernel<false>,
                        hipFuncAttributeMaxDynamicSharedMemorySize, SMEM_BYTES);

    moe_zero_kernel<<<2048, 256, 0, stream>>>(out, cnt);
    if (pre)
        moe_convert_kernel<<<4096, 256, 0, stream>>>(w_fc, w_proj, wfc_bf, wpj_bf);
    moe_router_kernel<<<T_ / 8, 256, 0, stream>>>(x, w_gate, logits, cnt, tlist, wlist);

    dim3 grid(T_ / TILE_M, E_);
    if (pre)
        moe_expert_kernel<true><<<grid, 256, SMEM_BYTES, stream>>>(
            x, w_fc, wfc_bf, b_fc, w_proj, wpj_bf, b_proj, cnt, tlist, wlist, out);
    else
        moe_expert_kernel<false><<<grid, 256, SMEM_BYTES, stream>>>(
            x, w_fc, wfc_bf, b_fc, w_proj, wpj_bf, b_proj, cnt, tlist, wlist, out);
}
